// NTN_81827716923973
// MI455X (gfx1250) — compile-verified, matrix operand order checked by reading, not run on hardware
//
#include <hip/hip_runtime.h>

typedef __attribute__((ext_vector_type(16))) _Float16 v16h;
typedef __attribute__((ext_vector_type(8)))  _Float16 v8h;
typedef __attribute__((ext_vector_type(8)))  float    v8f;
typedef __attribute__((ext_vector_type(4)))  uint32_t v4u;
typedef __attribute__((ext_vector_type(8)))  uint32_t v8u;

#define N_ROWS 8192
#define DDIM   1024
#define KDIM   64
#define BM     128   // x1 rows per block (8 waves x 16)
#define KC     32    // D-chunk per WMMA step (f16 K=32)
#define LDH    40    // padded LDS half-stride: 80B (TDM pad reproduces this)

// ---------------------------------------------------------------------------
// Kernel 1: fold x2 into W, fuse +V1, store M = V1 + W@x2 as f16.
// Pure streaming GEMV over 256 MB -> the dominant, bandwidth-bound pass
// (~11 us at 23.3 TB/s). Wave-per-row, b128 loads, x2 in LDS.
// ---------------------------------------------------------------------------
__global__ void ntn_fold_w(const float* __restrict__ W,
                           const float* __restrict__ V,
                           const float* __restrict__ x2,
                           _Float16* __restrict__ Mh) {
    __shared__ __align__(16) float sx2[DDIM];
    const int tid = threadIdx.x;
    #pragma unroll
    for (int i = 0; i < DDIM / 256; ++i) sx2[tid + 256 * i] = x2[tid + 256 * i];
    __syncthreads();

    const int lane = tid & 31;
    const int wave = tid >> 5;
    const int row  = blockIdx.x * 8 + wave;          // row in [0, K*D)
    const float* wp = W + (size_t)row * DDIM;

    float sum = 0.f;
    #pragma unroll
    for (int i = 0; i < DDIM / 128; ++i) {
        const int e = i * 128 + lane * 4;
        __builtin_prefetch(wp + e + 256, 0, 0);      // -> global_prefetch_b8
        float4 w4 = *(const float4*)(wp + e);
        float4 x4 = *(const float4*)(sx2 + e);
        sum += w4.x * x4.x + w4.y * x4.y + w4.z * x4.z + w4.w * x4.w;
    }
    #pragma unroll
    for (int off = 16; off >= 1; off >>= 1) sum += __shfl_xor(sum, off, 32);

    if (lane == 0) {
        const int k = row >> 10;
        const int d = row & (DDIM - 1);
        Mh[row] = (_Float16)(sum + V[(size_t)k * 2 * DDIM + d]);   // + V1[k,d]
    }
}

// ---------------------------------------------------------------------------
// Kernel 2: c[k] = V2[k] . x2 + b[k]   (tiny)
// ---------------------------------------------------------------------------
__global__ void ntn_cvec(const float* __restrict__ V,
                         const float* __restrict__ x2,
                         const float* __restrict__ b,
                         float* __restrict__ cbuf) {
    __shared__ float red[256];
    const int k = blockIdx.x, tid = threadIdx.x;
    const float* v2 = V + (size_t)k * 2 * DDIM + DDIM;
    float s = 0.f;
    for (int d = tid; d < DDIM; d += 256) s += v2[d] * x2[d];
    red[tid] = s;
    __syncthreads();
    #pragma unroll
    for (int st = 128; st >= 1; st >>= 1) {
        if (tid < st) red[tid] += red[tid + st];
        __syncthreads();
    }
    if (tid == 0) cbuf[k] = red[0] + b[k];
}

// ---------------------------------------------------------------------------
// Kernel 3: out[n] = sum_k U[k] * relu( x1[n] . M[k] + c[k] )
// WMMA f32<-f16 16x16x32. B tile (Mh) staged via the Tensor Data Mover with
// hardware LDS padding to the conflict-free 80B row stride; x1 tile staged
// through VGPRs because it needs f32->f16 conversion.
// ---------------------------------------------------------------------------
__global__ void ntn_gemm(const float* __restrict__ x1,
                         const _Float16* __restrict__ Mh,
                         const float* __restrict__ cbuf,
                         const float* __restrict__ U,
                         float* __restrict__ out) {
    __shared__ __align__(16) _Float16 sX[BM][LDH];     // x1 tile, f16
    __shared__ __align__(16) _Float16 sM[KDIM][LDH];   // M tile, TDM-filled

    const int tid  = threadIdx.x;
    const int lane = tid & 31;
    const int wave = tid >> 5;
    const int n    = lane & 15;     // row/col-in-tile index
    const int h    = lane >> 4;     // K-half select per ISA f16 layout
    const int rowBase = blockIdx.x * BM;

    // LDS byte offset of sM (generic shared address: LDS offset = addr[31:0])
    const uint32_t ldsM = (uint32_t)(uintptr_t)&sM[0][0];

    v8f acc[4] = {};

    for (int dBase = 0; dBase < DDIM; dBase += KC) {
        __syncthreads();   // previous iteration's fragment reads complete

        if (wave == 0) {
            // ---- Tensor DMA descriptor (2-D tensor -> groups 0+1 only) ----
            const uint64_t ga = (uint64_t)(uintptr_t)Mh + (uint64_t)dBase * 2;
            v4u g0 = {};
            g0[0] = 1u;                                    // count=1, user mode
            g0[1] = ldsM;                                  // lds_addr
            g0[2] = (uint32_t)ga;                          // global_addr[31:0]
            g0[3] = ((uint32_t)(ga >> 32) & 0x01FFFFFFu)   // global_addr[56:32]
                    | (2u << 30);                          // type=2 ("image")
            v8u g1 = {};
            g1[0] = (1u << 16)        // data_size = 2 bytes
                  | (1u << 20)        // pad_enable
                  | (3u << 22)        // pad_interval: 16 DWORDs (= one 64B row)
                  | (3u << 25);       // pad_amount:   4 DWORDs (-> 80B stride)
            g1[1] = (uint32_t)(DDIM & 0xFFFF) << 16;       // tensor_dim0=1024 lo16
            g1[2] = ((uint32_t)KDIM << 16);                // dim0 hi=0 | tensor_dim1=64
            g1[3] = ((uint32_t)KC << 16);                  // dim1 hi=0 | tile_dim0=32
            g1[4] = (uint32_t)KDIM;                        // tile_dim1=64, tile_dim2=0
            g1[5] = (uint32_t)DDIM;                        // tensor_dim0_stride=1024
            g1[6] = 0u;                                    // stride hi | dim1_stride lo
            g1[7] = 0u;
            asm volatile("tensor_load_to_lds %0, %1"
                         :: "s"(g0), "s"(g1) : "memory");
        }

        // stage 128x32 of x1 (f32 -> f16) while the TDM streams the B tile
        #pragma unroll
        for (int i = 0; i < 4; ++i) {
            const int q  = tid + 256 * i;          // float4 id in [0,1024)
            const int r  = q >> 3;
            const int c4 = (q & 7) * 4;
            float4 v = *(const float4*)(x1 + (size_t)(rowBase + r) * DDIM + dBase + c4);
            sX[r][c4 + 0] = (_Float16)v.x;  sX[r][c4 + 1] = (_Float16)v.y;
            sX[r][c4 + 2] = (_Float16)v.z;  sX[r][c4 + 3] = (_Float16)v.w;
        }

        if (wave == 0) __builtin_amdgcn_s_wait_tensorcnt(0);
        __syncthreads();   // publish sX stores + TDM-filled sM to all waves

        // A fragment: lane holds K = {h*8..h*8+7, 16+h*8..+7} of row n
        // -> two 16B ds_load_b128 from the padded tile.
        v8h alo = *(const v8h*)&sX[wave * 16 + n][h * 8];
        v8h ahi = *(const v8h*)&sX[wave * 16 + n][16 + h * 8];
        v16h a = __builtin_shufflevector(alo, ahi,
                 0,1,2,3,4,5,6,7,8,9,10,11,12,13,14,15);

        #pragma unroll
        for (int t = 0; t < 4; ++t) {
            v8h blo = *(const v8h*)&sM[t * 16 + n][h * 8];
            v8h bhi = *(const v8h*)&sM[t * 16 + n][16 + h * 8];
            v16h bb = __builtin_shufflevector(blo, bhi,
                     0,1,2,3,4,5,6,7,8,9,10,11,12,13,14,15);
            acc[t] = __builtin_amdgcn_wmma_f32_16x16x32_f16(
                         false, a, false, bb, (short)0, acc[t], false, false);
        }
    }

    // Fused epilogue: relu(acc + c) * U, reduce over the 64 columns.
    // C layout: lane,reg r -> (M = r + 8*h, N = t*16 + n); xor 1,2,4,8 stays
    // within each 16-lane half-group, summing over N.
    float rowsum[8];
    #pragma unroll
    for (int r = 0; r < 8; ++r) rowsum[r] = 0.f;

    #pragma unroll
    for (int t = 0; t < 4; ++t) {
        const float cv = cbuf[t * 16 + n];
        const float uu = U[t * 16 + n];
        #pragma unroll
        for (int r = 0; r < 8; ++r) {
            float v = acc[t][r] + cv;
            v = fmaxf(v, 0.f) * uu;
            v += __shfl_xor(v, 1, 32);
            v += __shfl_xor(v, 2, 32);
            v += __shfl_xor(v, 4, 32);
            v += __shfl_xor(v, 8, 32);
            rowsum[r] += v;
        }
    }
    if (n == 0) {
        #pragma unroll
        for (int r = 0; r < 8; ++r)
            out[rowBase + wave * 16 + r + 8 * h] = rowsum[r];
    }
}

// ---------------------------------------------------------------------------
extern "C" void kernel_launch(void* const* d_in, const int* in_sizes, int n_in,
                              void* d_out, int out_size, void* d_ws, size_t ws_size,
                              hipStream_t stream) {
    const float* x1 = (const float*)d_in[0];   // (8192, 1024)
    const float* x2 = (const float*)d_in[1];   // (1, 1024)
    const float* V  = (const float*)d_in[2];   // (64, 2048)
    const float* W  = (const float*)d_in[3];   // (64, 1024, 1024)
    const float* b  = (const float*)d_in[4];   // (64,)
    const float* U  = (const float*)d_in[5];   // (64, 1)
    float* out = (float*)d_out;                // (8192, 1)

    _Float16* Mh = (_Float16*)d_ws;                                   // 128 KB
    float* cbuf  = (float*)((char*)d_ws + (size_t)KDIM * DDIM * 2);   // 256 B

    ntn_fold_w<<<dim3((KDIM * DDIM) / 8), 256, 0, stream>>>(W, V, x2, Mh);
    ntn_cvec  <<<dim3(KDIM),              256, 0, stream>>>(V, x2, b, cbuf);
    ntn_gemm  <<<dim3(N_ROWS / BM),       256, 0, stream>>>(x1, Mh, cbuf, U, out);
}
